// TensorVMSplit_73985106640954
// MI455X (gfx1250) — compile-verified, matrix operand order checked by reading, not run on hardware
//
#include <hip/hip_runtime.h>
#include <hip/hip_bf16.h>
#include <math.h>

typedef _Float16 half_t;
typedef __attribute__((ext_vector_type(16))) _Float16 v16h;
typedef __attribute__((ext_vector_type(8)))  _Float16 v8h;
typedef __attribute__((ext_vector_type(8)))  float    v8f;

#define GRID_N   300
#define NSAMP    512
#define STEPSZ   (3.0f / (GRID_N - 1) * 0.5f)
#define DISTSC   25.0f
#define DSHIFT   (-10.0f)
#define WTHRESH  1e-4f
#define CD       16
#define CC       16
#define DFEAT    27
#define FIN      150
#define FINP     160   // padded K for layer 1
#define FHID     128

// ---- dynamic LDS partition (bytes) ----
constexpr int OFF_W1  = 0;                          // 128x160 f16
constexpr int OFF_W2  = OFF_W1  + FHID * FINP * 2;  // 40960: 128x128 f16
constexpr int OFF_BAS = OFF_W2  + FHID * FHID * 2;  // 73728: 27x48 f32
constexpr int OFF_B1  = OFF_BAS + DFEAT * 48 * 4;   // 78912
constexpr int OFF_B2  = OFF_B1  + FHID * 4;         // 79424
constexpr int OFF_W3  = OFF_B2  + FHID * 4;         // 79936: 3x128 f32
constexpr int OFF_B3  = OFF_W3  + 3 * FHID * 4;     // 81472 (pad to 16)
constexpr int OFF_SIG = OFF_B3  + 16;               // 81488: 512 f32
constexpr int OFF_RGB = OFF_SIG + NSAMP * 4;        // 83536: 512x3 f32
constexpr int OFF_A   = OFF_RGB + NSAMP * 3 * 4;    // 89680: 8 waves x 32 x 160 f16
constexpr int OFF_HA  = OFF_A   + 8 * 32 * FINP * 2;// 171600: 8 x 32 x 128 f16
constexpr int SMEM_TOTAL = OFF_HA + 8 * 32 * FHID * 2; // 237136 B (< 320 KB/WG)
// NOTE: the layer-2 output buffer (hB, 32x128 f16 per wave) ALIASES the wave's
// A-tile region: A is fully register-resident before layer-1 issues, so its
// LDS region is dead by the time layer-2 writes.

__device__ __forceinline__ void wave_lds_fence() {
  asm volatile("s_wait_dscnt 0x0" ::: "memory");
}

// Load a 16x32 f16 A/B operand fragment from an LDS row (row-major, contiguous K).
// Layout per ISA 7.12.2: lanes 0-15 hold K = ks+0..7 and ks+16..23,
// lanes 16-31 hold K = ks+8..15 and ks+24..31 (hi = lane>>4).
__device__ __forceinline__ v16h load_frag(const half_t* __restrict__ row, int hi, int ks) {
  v8h lo = *(const v8h*)(row + ks + hi * 8);
  v8h hh = *(const v8h*)(row + ks + 16 + hi * 8);
  v16h r;
#pragma unroll
  for (int i = 0; i < 8; ++i) { r[i] = lo[i]; r[i + 8] = hh[i]; }
  return r;
}

__device__ __forceinline__ v8f wmma_f16(v16h a, v16h b, v8f c) {
  return __builtin_amdgcn_wmma_f32_16x16x32_f16(false, a, false, b,
                                                (short)0, c, false, false);
}

__device__ __forceinline__ void plane_taps(float px, float py, int idx[4], float w[4]) {
  float fx = (px + 1.0f) * 0.5f * (GRID_N - 1);
  float fy = (py + 1.0f) * 0.5f * (GRID_N - 1);
  float x0f = floorf(fx), y0f = floorf(fy);
  float wx = fx - x0f, wy = fy - y0f;
  int x0 = (int)x0f, y0 = (int)y0f;
  int x1 = x0 + 1, y1 = y0 + 1;
  float vx0 = (x0 >= 0 && x0 < GRID_N) ? 1.f : 0.f;
  float vx1 = (x1 >= 0 && x1 < GRID_N) ? 1.f : 0.f;
  float vy0 = (y0 >= 0 && y0 < GRID_N) ? 1.f : 0.f;
  float vy1 = (y1 >= 0 && y1 < GRID_N) ? 1.f : 0.f;
  int cx0 = min(max(x0, 0), GRID_N - 1), cx1 = min(max(x1, 0), GRID_N - 1);
  int cy0 = min(max(y0, 0), GRID_N - 1), cy1 = min(max(y1, 0), GRID_N - 1);
  idx[0] = cy0 * GRID_N + cx0; w[0] = (1.f - wx) * (1.f - wy) * vx0 * vy0;
  idx[1] = cy0 * GRID_N + cx1; w[1] = wx * (1.f - wy) * vx1 * vy0;
  idx[2] = cy1 * GRID_N + cx0; w[2] = (1.f - wx) * wy * vx0 * vy1;
  idx[3] = cy1 * GRID_N + cx1; w[3] = wx * wy * vx1 * vy1;
}

__device__ __forceinline__ void line_taps(float pz, int idx[2], float w[2]) {
  float fz = (pz + 1.0f) * 0.5f * (GRID_N - 1);
  float z0f = floorf(fz);
  float wz = fz - z0f;
  int z0 = (int)z0f, z1 = z0 + 1;
  float v0 = (z0 >= 0 && z0 < GRID_N) ? 1.f : 0.f;
  float v1 = (z1 >= 0 && z1 < GRID_N) ? 1.f : 0.f;
  idx[0] = min(max(z0, 0), GRID_N - 1); w[0] = (1.f - wz) * v0;
  idx[1] = min(max(z1, 0), GRID_N - 1); w[1] = wz * v1;
}

__launch_bounds__(256, 1)
__global__ void tensorf_fused_kernel(
    const float* __restrict__ rays,         // (R,6)
    const float* __restrict__ dense_plane,  // (3,16,300,300)
    const float* __restrict__ dense_line,   // (3,16,300)
    const float* __restrict__ color_plane,  // (3,16,300,300)
    const float* __restrict__ color_line,   // (3,16,300)
    const float* __restrict__ basis,        // (27,48)
    const float* __restrict__ w1, const float* __restrict__ b1,  // (128,150),(128)
    const float* __restrict__ w2, const float* __restrict__ b2,  // (128,128),(128)
    const float* __restrict__ w3, const float* __restrict__ b3,  // (3,128),(3)
    float* __restrict__ out, int R)
{
  extern __shared__ __align__(16) char smem[];
  half_t* sW1  = (half_t*)(smem + OFF_W1);
  half_t* sW2  = (half_t*)(smem + OFF_W2);
  float*  sBas = (float*)(smem + OFF_BAS);
  float*  sB1  = (float*)(smem + OFF_B1);
  float*  sB2  = (float*)(smem + OFF_B2);
  float*  sW3  = (float*)(smem + OFF_W3);
  float*  sB3  = (float*)(smem + OFF_B3);
  float*  sSig = (float*)(smem + OFF_SIG);
  float*  sRGB = (float*)(smem + OFF_RGB);
  half_t* sA   = (half_t*)(smem + OFF_A);
  half_t* sHA  = (half_t*)(smem + OFF_HA);

  const int tid  = threadIdx.x;
  const int wave = tid >> 5;
  const int lane = tid & 31;
  const int r    = blockIdx.x;

  // ---- stage weights (f32 -> f16 where matrices feed WMMA) ----
  for (int i = tid; i < FHID * FINP; i += 256) {
    int n = i / FINP, k = i % FINP;
    sW1[i] = (half_t)((k < FIN) ? w1[n * FIN + k] : 0.f);
  }
  for (int i = tid; i < FHID * FHID; i += 256) sW2[i] = (half_t)w2[i];
  for (int i = tid; i < DFEAT * 48; i += 256)  sBas[i] = basis[i];
  for (int i = tid; i < FHID; i += 256) { sB1[i] = b1[i]; sB2[i] = b2[i]; }
  for (int i = tid; i < 3 * FHID; i += 256) sW3[i] = w3[i];
  if (tid < 3) sB3[tid] = b3[tid];
  __syncthreads();

  // ---- ray setup (every thread recomputes; cheap) ----
  const float ox = rays[r * 6 + 0], oy = rays[r * 6 + 1], oz = rays[r * 6 + 2];
  const float dx = rays[r * 6 + 3], dy = rays[r * 6 + 4], dz = rays[r * 6 + 5];
  const float vx = (dx == 0.f) ? 1e-6f : dx;
  const float vy = (dy == 0.f) ? 1e-6f : dy;
  const float vz = (dz == 0.f) ? 1e-6f : dz;
  float tx = fminf((1.5f - ox) / vx, (-1.5f - ox) / vx);
  float ty = fminf((1.5f - oy) / vy, (-1.5f - oy) / vy);
  float tz = fminf((1.5f - oz) / vz, (-1.5f - oz) / vz);
  float tmin = fminf(fmaxf(fmaxf(fmaxf(tx, ty), tz), 2.0f), 6.0f);

  const int matx[3] = {0, 0, 1};
  const int maty[3] = {1, 2, 2};
  const int vecz[3] = {2, 1, 0};

  half_t* Aw  = sA  + wave * 32 * FINP;         // 32 x 160 f16 (A tile)
  half_t* hAw = sHA + wave * 32 * FHID;         // 32 x 128 f16 (layer-1 out)
  half_t* hBw = Aw;                             // 32 x 128 f16, ALIASES A tile
  const int hi = lane >> 4, nloc = lane & 15;

  for (int g = 0; g < 2; ++g) {
    const int sbase = wave * 64 + g * 32;
    const int s = sbase + lane;

    // ================= feature phase: one sample per lane =================
    {
      float z = tmin + STEPSZ * (float)s;
      float px_ = ox + dx * z, py_ = oy + dy * z, pz_ = oz + dz * z;
      bool valid = (px_ >= -1.5f && px_ <= 1.5f) &&
                   (py_ >= -1.5f && py_ <= 1.5f) &&
                   (pz_ >= -1.5f && pz_ <= 1.5f);
      float xyz[3] = {px_ * (2.f / 3.f), py_ * (2.f / 3.f), pz_ * (2.f / 3.f)};

      float sf = 0.f;
      float app[DFEAT];
#pragma unroll
      for (int i = 0; i < DFEAT; ++i) app[i] = 0.f;

#pragma unroll
      for (int ax = 0; ax < 3; ++ax) {
        int   pidx[4]; float pw[4];
        int   lidx[2]; float lw[2];
        plane_taps(xyz[matx[ax]], xyz[maty[ax]], pidx, pw);
        line_taps(xyz[vecz[ax]], lidx, lw);
        const float* dpb = dense_plane + ax * CD * GRID_N * GRID_N;
        const float* dlb = dense_line  + ax * CD * GRID_N;
        const float* cpb = color_plane + ax * CC * GRID_N * GRID_N;
        const float* clb = color_line  + ax * CC * GRID_N;
        for (int c = 0; c < CD; ++c) {
          const float* P = dpb + c * (GRID_N * GRID_N);
          float pv = pw[0] * P[pidx[0]] + pw[1] * P[pidx[1]] +
                     pw[2] * P[pidx[2]] + pw[3] * P[pidx[3]];
          const float* L = dlb + c * GRID_N;
          float lv = lw[0] * L[lidx[0]] + lw[1] * L[lidx[1]];
          sf += pv * lv;

          const float* Pc = cpb + c * (GRID_N * GRID_N);
          float pvc = pw[0] * Pc[pidx[0]] + pw[1] * Pc[pidx[1]] +
                      pw[2] * Pc[pidx[2]] + pw[3] * Pc[pidx[3]];
          const float* Lc = clb + c * GRID_N;
          float lvc = lw[0] * Lc[lidx[0]] + lw[1] * Lc[lidx[1]];
          float pr = pvc * lvc;
          int j = ax * CC + c;
#pragma unroll
          for (int i = 0; i < DFEAT; ++i) app[i] += pr * sBas[i * 48 + j];
        }
      }

      // sigma = valid ? softplus(sf - 10) : 0
      float sp = sf + DSHIFT;
      float sg = (sp > 20.f) ? sp : log1pf(__expf(sp));
      sSig[s] = valid ? sg : 0.f;

      // mlp_in row -> LDS (f16), padded to 160
      half_t* Arow = Aw + lane * FINP;
#pragma unroll
      for (int i = 0; i < DFEAT; ++i) Arow[i] = (half_t)app[i];
      Arow[27] = (half_t)dx; Arow[28] = (half_t)dy; Arow[29] = (half_t)dz;
#pragma unroll
      for (int i = 0; i < DFEAT; ++i) {
        Arow[30 + 2 * i]     = (half_t)__sinf(app[i]);
        Arow[30 + 2 * i + 1] = (half_t)__sinf(2.f * app[i]);
        Arow[84 + 2 * i]     = (half_t)__cosf(app[i]);
        Arow[84 + 2 * i + 1] = (half_t)__cosf(2.f * app[i]);
      }
      float dv[3] = {dx, dy, dz};
#pragma unroll
      for (int i = 0; i < 3; ++i) {
        Arow[138 + 2 * i]     = (half_t)__sinf(dv[i]);
        Arow[138 + 2 * i + 1] = (half_t)__sinf(2.f * dv[i]);
        Arow[144 + 2 * i]     = (half_t)__cosf(dv[i]);
        Arow[144 + 2 * i + 1] = (half_t)__cosf(2.f * dv[i]);
      }
#pragma unroll
      for (int k = FIN; k < FINP; ++k) Arow[k] = (half_t)0.f;
    }
    wave_lds_fence();  // same-wave LDS stores -> loads ordering

    // ======== MLP: both 16-row subtiles per pass, A register-resident ========
    // ---- hoist layer-1 A fragments (2 subtiles x 5 K-steps) ----
    v16h a1f[2][5];
#pragma unroll
    for (int sub = 0; sub < 2; ++sub) {
      const half_t* Arow = Aw + (sub * 16 + nloc) * FINP;
#pragma unroll
      for (int ks = 0; ks < 5; ++ks) a1f[sub][ks] = load_frag(Arow, hi, ks * 32);
    }
    wave_lds_fence();  // A region (aliased by hB) must be fully read

    // ---- layer 1: 32x160 @ 160x128, B loaded once per (nt,ks) ----
    for (int nt = 0; nt < 8; ++nt) {
      const half_t* Brow = sW1 + (nt * 16 + nloc) * FINP;
      v8f acc0 = {}, acc1 = {};
#pragma unroll
      for (int ks = 0; ks < 5; ++ks) {
        v16h b = load_frag(Brow, hi, ks * 32);
        acc0 = wmma_f16(a1f[0][ks], b, acc0);
        acc1 = wmma_f16(a1f[1][ks], b, acc1);
      }
      float bias = sB1[nt * 16 + nloc];
#pragma unroll
      for (int j = 0; j < 8; ++j) {
        float v0 = fmaxf(acc0[j] + bias, 0.f);
        float v1 = fmaxf(acc1[j] + bias, 0.f);
        hAw[(j + 8 * hi) * FHID + nt * 16 + nloc]        = (half_t)v0;
        hAw[(16 + j + 8 * hi) * FHID + nt * 16 + nloc]   = (half_t)v1;
      }
    }
    wave_lds_fence();

    // ---- layer 2: 32x128 @ 128x128 ----
    v16h a2f[2][4];
#pragma unroll
    for (int sub = 0; sub < 2; ++sub) {
      const half_t* Hrow = hAw + (sub * 16 + nloc) * FHID;
#pragma unroll
      for (int ks = 0; ks < 4; ++ks) a2f[sub][ks] = load_frag(Hrow, hi, ks * 32);
    }
    for (int nt = 0; nt < 8; ++nt) {
      const half_t* Brow = sW2 + (nt * 16 + nloc) * FHID;
      v8f acc0 = {}, acc1 = {};
#pragma unroll
      for (int ks = 0; ks < 4; ++ks) {
        v16h b = load_frag(Brow, hi, ks * 32);
        acc0 = wmma_f16(a2f[0][ks], b, acc0);
        acc1 = wmma_f16(a2f[1][ks], b, acc1);
      }
      float bias = sB2[nt * 16 + nloc];
#pragma unroll
      for (int j = 0; j < 8; ++j) {
        float v0 = fmaxf(acc0[j] + bias, 0.f);
        float v1 = fmaxf(acc1[j] + bias, 0.f);
        hBw[(j + 8 * hi) * FHID + nt * 16 + nloc]        = (half_t)v0;
        hBw[(16 + j + 8 * hi) * FHID + nt * 16 + nloc]   = (half_t)v1;
      }
    }
    wave_lds_fence();

    // ---- layer 3: 128 -> 3, sigmoid (VALU; all 32 lanes, one sample each) ----
    {
      const half_t* hrow = hBw + lane * FHID;
      float a0 = sB3[0], a1 = sB3[1], a2 = sB3[2];
#pragma unroll
      for (int kc = 0; kc < FHID / 8; ++kc) {
        v8h h8 = *(const v8h*)(hrow + kc * 8);
#pragma unroll
        for (int t = 0; t < 8; ++t) {
          float hv = (float)h8[t];
          int k = kc * 8 + t;
          a0 += sW3[0 * FHID + k] * hv;
          a1 += sW3[1 * FHID + k] * hv;
          a2 += sW3[2 * FHID + k] * hv;
        }
      }
      sRGB[s * 3 + 0] = 1.f / (1.f + __expf(-a0));
      sRGB[s * 3 + 1] = 1.f / (1.f + __expf(-a1));
      sRGB[s * 3 + 2] = 1.f / (1.f + __expf(-a2));
    }
    wave_lds_fence();  // hB (== A region) reused by next group's feature phase
  }

  __syncthreads();

  // ================= transmittance scan + composition (lane 0) =================
  if (tid == 0) {
    float T = 1.f, acc = 0.f, dep = 0.f, r0 = 0.f, r1 = 0.f, r2 = 0.f;
    for (int s2 = 0; s2 < NSAMP; ++s2) {
      float sg = sSig[s2];
      float dist = (s2 == NSAMP - 1) ? 0.f : STEPSZ;
      float alpha = 1.f - __expf(-sg * dist * DISTSC);
      float w = alpha * T;
      T *= (1.f - alpha + 1e-10f);
      acc += w;
      dep += w * (tmin + STEPSZ * (float)s2);
      if (w > WTHRESH) {
        r0 += w * sRGB[s2 * 3 + 0];
        r1 += w * sRGB[s2 * 3 + 1];
        r2 += w * sRGB[s2 * 3 + 2];
      }
    }
    float bg = 1.f - acc;
    out[r * 3 + 0] = fminf(fmaxf(r0 + bg, 0.f), 1.f);
    out[r * 3 + 1] = fminf(fmaxf(r1 + bg, 0.f), 1.f);
    out[r * 3 + 2] = fminf(fmaxf(r2 + bg, 0.f), 1.f);
    out[R * 3 + r] = dep;
  }
}

extern "C" void kernel_launch(void* const* d_in, const int* in_sizes, int n_in,
                              void* d_out, int out_size, void* d_ws, size_t ws_size,
                              hipStream_t stream) {
  const float* rays  = (const float*)d_in[0];
  const float* dpln  = (const float*)d_in[1];
  const float* dlin  = (const float*)d_in[2];
  const float* cpln  = (const float*)d_in[3];
  const float* clin  = (const float*)d_in[4];
  const float* basis = (const float*)d_in[5];
  const float* w1    = (const float*)d_in[6];
  const float* b1    = (const float*)d_in[7];
  const float* w2    = (const float*)d_in[8];
  const float* b2    = (const float*)d_in[9];
  const float* w3    = (const float*)d_in[10];
  const float* b3    = (const float*)d_in[11];
  // d_in[12] = n_samples (512, fixed by kernel structure)

  int R = in_sizes[0] / 6;
  float* out = (float*)d_out;

  (void)d_ws; (void)ws_size; (void)n_in; (void)out_size;

  hipFuncSetAttribute((const void*)tensorf_fused_kernel,
                      hipFuncAttributeMaxDynamicSharedMemorySize, SMEM_TOTAL);
  tensorf_fused_kernel<<<dim3(R), dim3(256), SMEM_TOTAL, stream>>>(
      rays, dpln, dlin, cpln, clin, basis, w1, b1, w2, b2, w3, b3, out, R);
}
